// MoDTransformerBlock_40192303956200
// MI455X (gfx1250) — matrix operand order repaired and checked by reading, not compiled, not run on hardware
//
#include <hip/hip_runtime.h>
#include <hip/hip_bf16.h>
#include <cstdint>

// ---------------- problem constants ----------------
#define NB     4
#define TSEQ   4096
#define DMODEL 1024
#define NHEAD  16
#define HDIM   64
#define CAPK   512
#define NTOK   (NB * CAPK)   // 2048
#define D3     3072
#define DFF    2730
#define DFFP   2816          // DFF padded to multiple of 128

typedef __attribute__((ext_vector_type(16))) __bf16 v16bf;
typedef __attribute__((ext_vector_type(8)))  float  v8f;

// ---------------- scalar bf16 helpers ----------------
static __device__ __forceinline__ unsigned short f2bf(float f) {
  union { float f; unsigned u; } c; c.f = f;
  unsigned u = c.u;
  unsigned r = (u + 0x7FFFu + ((u >> 16) & 1u)) >> 16;   // RNE
  return (unsigned short)r;
}

// ---------------- WMMA helpers (CDNA5, wave32) ----------------
static __device__ __forceinline__ v8f wmma_bf16(v16bf a, v16bf b, v8f c) {
  // D = A(16x32 bf16) * B(32x16 bf16) + C(16x16 f32)
  return __builtin_amdgcn_wmma_f32_16x16x32_bf16(false, a, false, b, (short)0, c,
                                                 false, false);
}

// A fragment: 16(M) x 32(K) bf16, source row-major with leading dim `ld` (elems).
// ISA layout: lane L -> M = L%16 ; half h=L/16 ; v0..3: K=8h+2v,{+1} ; v4..7: K=16+8h+2(v-4),{+1}
static __device__ __forceinline__ v16bf load_a_frag(const unsigned short* base, int ld) {
  const int l = threadIdx.x & 31;
  const int row = l & 15, h = l >> 4;
  const unsigned short* p = base + (size_t)row * ld;
  union { v16bf v; unsigned d[8]; } u;
#pragma unroll
  for (int i = 0; i < 4; ++i) u.d[i]     = *(const unsigned*)(p + 8 * h + 2 * i);
#pragma unroll
  for (int i = 0; i < 4; ++i) u.d[4 + i] = *(const unsigned*)(p + 16 + 8 * h + 2 * i);
  return u.v;
}

// B fragment: 32(K) x 16(N) bf16 where source is stored N-major: B[k][n] = src[n*ld + k].
// ISA layout: lane L -> N = L%16 ; half h=L/16 ; v0..3: K=8h+2v,{+1} ; v4..7: K=16+8h+2(v-4),{+1}
static __device__ __forceinline__ v16bf load_b_frag(const unsigned short* base, int ld) {
  const int l = threadIdx.x & 31;
  const int n = l & 15, h = l >> 4;
  const unsigned short* p = base + (size_t)n * ld;
  union { v16bf v; unsigned d[8]; } u;
#pragma unroll
  for (int i = 0; i < 4; ++i) u.d[i]     = *(const unsigned*)(p + 8 * h + 2 * i);
#pragma unroll
  for (int i = 0; i < 4; ++i) u.d[4 + i] = *(const unsigned*)(p + 16 + 8 * h + 2 * i);
  return u.v;
}

// ---------------- block reduce (256 threads) ----------------
static __device__ __forceinline__ float block_reduce_sum256(float v) {
  __shared__ float sh[9];
#pragma unroll
  for (int m = 16; m >= 1; m >>= 1) v += __shfl_xor(v, m, 32);
  if ((threadIdx.x & 31) == 0) sh[threadIdx.x >> 5] = v;
  __syncthreads();
  if (threadIdx.x == 0) {
    float t = 0.f;
#pragma unroll
    for (int i = 0; i < 8; ++i) t += sh[i];
    sh[8] = t;
  }
  __syncthreads();
  return sh[8];
}

// ---------------- 1) router scores: s[b,t] = x[b,t,:] . w ----------------
__global__ __launch_bounds__(256) void router_kernel(const float* __restrict__ x,
                                                     const float* __restrict__ w,
                                                     float* __restrict__ sc) {
  const int tok = blockIdx.x * 8 + (threadIdx.x >> 5);
  const int l = threadIdx.x & 31;
  const float* xr = x + (size_t)tok * DMODEL;
  float s = 0.f;
  for (int c = l; c < DMODEL; c += 32) s += xr[c] * w[c];
#pragma unroll
  for (int m = 16; m >= 1; m >>= 1) s += __shfl_xor(s, m, 32);
  if (l == 0) sc[tok] = s;
}

// ---------------- 2) deterministic top-k by rank (desc score, stable ties) ----------------
__global__ __launch_bounds__(256) void topk_kernel(const float* __restrict__ sc,
                                                   int* __restrict__ idx) {
  __shared__ float s[TSEQ];
  const int b = blockIdx.x;
  for (int i = threadIdx.x; i < TSEQ; i += 256) s[i] = sc[b * TSEQ + i];
  __syncthreads();
  for (int i = threadIdx.x; i < TSEQ; i += 256) {
    const float si = s[i];
    int rank = 0;
    for (int j = 0; j < TSEQ; ++j) {
      const float sj = s[j];
      rank += (sj > si) || (sj == si && j < i);
    }
    if (rank < CAPK) idx[b * CAPK + rank] = i;
  }
}

// ---------------- 3) gather + RMSNorm -> x_sel (f32) and h (bf16) ----------------
__global__ __launch_bounds__(256) void gather_norm_kernel(const float* __restrict__ x,
                                                          const int* __restrict__ idx,
                                                          const float* __restrict__ g,
                                                          float* __restrict__ xsel,
                                                          unsigned short* __restrict__ hout) {
  const int b = blockIdx.y, j = blockIdx.x;
  const int t = idx[b * CAPK + j];
  const float* xr = x + ((size_t)b * TSEQ + t) * DMODEL;
  float ss = 0.f;
  for (int c = threadIdx.x; c < DMODEL; c += 256) { float v = xr[c]; ss += v * v; }
  ss = block_reduce_sum256(ss);
  const float inv = rsqrtf(ss * (1.0f / DMODEL) + 1e-6f);
  const size_t o = ((size_t)(b * CAPK + j)) * DMODEL;
  for (int c = threadIdx.x; c < DMODEL; c += 256) {
    const float v = xr[c];
    xsel[o + c] = v;
    hout[o + c] = f2bf(v * inv * g[c]);
  }
}

// ---------------- 4) RMSNorm of (x_sel + x_attn) -> h2 (bf16) ----------------
__global__ __launch_bounds__(256) void norm2_kernel(const float* __restrict__ xsel,
                                                    const float* __restrict__ xattn,
                                                    const float* __restrict__ g,
                                                    unsigned short* __restrict__ hout) {
  const size_t o = (size_t)(blockIdx.y * CAPK + blockIdx.x) * DMODEL;
  float ss = 0.f;
  for (int c = threadIdx.x; c < DMODEL; c += 256) {
    const float v = xsel[o + c] + xattn[o + c];
    ss += v * v;
  }
  ss = block_reduce_sum256(ss);
  const float inv = rsqrtf(ss * (1.0f / DMODEL) + 1e-6f);
  for (int c = threadIdx.x; c < DMODEL; c += 256) {
    const float v = xsel[o + c] + xattn[o + c];
    hout[o + c] = f2bf(v * inv * g[c]);
  }
}

// ---------------- 5) f32 -> bf16 weight convert with zero padding ----------------
__global__ __launch_bounds__(256) void cvt_pad_kernel(const float* __restrict__ in,
                                                      unsigned short* __restrict__ out,
                                                      int R, int C, int total, int Cp) {
  const int i = blockIdx.x * 256 + threadIdx.x;
  if (i >= total) return;
  const int r = i / Cp, c = i % Cp;
  out[i] = (r < R && c < C) ? f2bf(in[(size_t)r * C + c]) : (unsigned short)0;
}

// ---------------- 6) tiled bf16 WMMA GEMM: C[M,N] = A[M,K] * W[N,K]^T ----------------
// Block 256 threads = 8 waves; block tile 128x128, K-step 32; wave tile 64x32.
__global__ __launch_bounds__(256) void gemm_bf16_nt(const unsigned short* __restrict__ A,
                                                    const unsigned short* __restrict__ W,
                                                    float* __restrict__ C,
                                                    int M, int N, int K) {
  __shared__ unsigned short sA[128][40];   // +8 pad keeps 16B alignment, avoids conflicts
  __shared__ unsigned short sB[128][40];
  const int m0 = blockIdx.y * 128;
  const int n0 = blockIdx.x * 128;
  const int wave = threadIdx.x >> 5;
  const int wm = (wave >> 2) * 64;
  const int wn = (wave & 3) * 32;
  const int lrow = threadIdx.x >> 1;            // 0..127
  const int lcol = (threadIdx.x & 1) * 16;      // 0 or 16

  v8f acc[4][2];
#pragma unroll
  for (int m = 0; m < 4; ++m)
#pragma unroll
    for (int n = 0; n < 2; ++n)
#pragma unroll
      for (int r = 0; r < 8; ++r) acc[m][n][r] = 0.f;

  for (int k0 = 0; k0 < K; k0 += 32) {
    const uint4* ga = (const uint4*)(A + (size_t)(m0 + lrow) * K + k0 + lcol);
    const uint4* gb = (const uint4*)(W + (size_t)(n0 + lrow) * K + k0 + lcol);
    const uint4 va0 = ga[0], va1 = ga[1];
    const uint4 vb0 = gb[0], vb1 = gb[1];
    if (k0 + 32 < K) {
      __builtin_prefetch(A + (size_t)(m0 + lrow) * K + k0 + 32 + lcol, 0, 1);
      __builtin_prefetch(W + (size_t)(n0 + lrow) * K + k0 + 32 + lcol, 0, 1);
    }
    *(uint4*)&sA[lrow][lcol] = va0; *(uint4*)&sA[lrow][lcol + 8] = va1;
    *(uint4*)&sB[lrow][lcol] = vb0; *(uint4*)&sB[lrow][lcol + 8] = vb1;
    __syncthreads();

    v16bf bf0 = load_b_frag(&sB[wn][0], 40);
    v16bf bf1 = load_b_frag(&sB[wn + 16][0], 40);
#pragma unroll
    for (int m = 0; m < 4; ++m) {
      v16bf af = load_a_frag(&sA[wm + m * 16][0], 40);
      acc[m][0] = wmma_bf16(af, bf0, acc[m][0]);
      acc[m][1] = wmma_bf16(af, bf1, acc[m][1]);
    }
    __syncthreads();
  }

  const int hl = (threadIdx.x & 31) >> 4;
  const int ln = threadIdx.x & 15;
#pragma unroll
  for (int m = 0; m < 4; ++m)
#pragma unroll
    for (int n = 0; n < 2; ++n)
#pragma unroll
      for (int r = 0; r < 8; ++r) {
        const int row = m0 + wm + m * 16 + r + 8 * hl;   // C/D layout: M = vgpr + 8*(lane/16)
        const int col = n0 + wn + n * 16 + ln;
        C[(size_t)row * N + col] = acc[m][n][r];
      }
}

// ---------------- 7) split qkv f32 -> q,k (bf16 [B,H,CAP,HD]) and v^T (bf16 [B,H,HD,CAP]) ----
__global__ __launch_bounds__(256) void qkv_split_kernel(const float* __restrict__ qkv,
                                                        unsigned short* __restrict__ q,
                                                        unsigned short* __restrict__ k,
                                                        unsigned short* __restrict__ vt) {
  const int i = blockIdx.x * 256 + threadIdx.x;   // < NTOK*D3
  if (i >= NTOK * D3) return;
  const int tok = i / D3, col = i % D3;
  const int b = tok >> 9, r = tok & 511;
  const int three = col >> 10, rem = col & 1023;
  const int hh = rem >> 6, hd = rem & 63;
  const unsigned short v = f2bf(qkv[i]);
  const size_t bh = (size_t)(b * NHEAD + hh);
  if (three == 0)      q[(bh * CAPK + r) * HDIM + hd] = v;
  else if (three == 1) k[(bh * CAPK + r) * HDIM + hd] = v;
  else                 vt[(bh * HDIM + hd) * CAPK + r] = v;
}

// ---------------- 8) causal flash attention (gathered order), WMMA bf16 ----------------
// grid (8 q-blocks, H, B), 128 threads = 4 waves, 16 query rows per wave.
__global__ __launch_bounds__(128) void attn_kernel(const unsigned short* __restrict__ qb,
                                                   const unsigned short* __restrict__ kb,
                                                   const unsigned short* __restrict__ vtb,
                                                   unsigned short* __restrict__ ob) {
  __shared__ unsigned short sP[4][16][40];        // per-wave P tile, bf16
  const int wave = threadIdx.x >> 5;
  const int lane = threadIdx.x & 31;
  const int hl = lane >> 4, ln = lane & 15;
  const int b = blockIdx.z, hh = blockIdx.y;
  const int q0 = blockIdx.x * 64 + wave * 16;
  const size_t bh = (size_t)(b * NHEAD + hh);
  const unsigned short* Q  = qb  + bh * CAPK * HDIM;
  const unsigned short* Kp = kb  + bh * CAPK * HDIM;
  const unsigned short* Vt = vtb + bh * HDIM * CAPK;

  v16bf qf0 = load_a_frag(Q + (size_t)q0 * HDIM, HDIM);
  v16bf qf1 = load_a_frag(Q + (size_t)q0 * HDIM + 32, HDIM);

  v8f oacc[4];
  float mi[8], li[8];
#pragma unroll
  for (int nt = 0; nt < 4; ++nt)
#pragma unroll
    for (int r = 0; r < 8; ++r) oacc[nt][r] = 0.f;
#pragma unroll
  for (int r = 0; r < 8; ++r) { mi[r] = -1e30f; li[r] = 0.f; }

  for (int j0 = 0; j0 < q0 + 16; j0 += 32) {      // causal: tiles up to diagonal
    v8f s0, s1;
#pragma unroll
    for (int r = 0; r < 8; ++r) { s0[r] = 0.f; s1[r] = 0.f; }
    {
      v16bf kf0 = load_b_frag(Kp + (size_t)j0 * HDIM, HDIM);
      v16bf kf1 = load_b_frag(Kp + (size_t)(j0 + 16) * HDIM, HDIM);
      s0 = wmma_bf16(qf0, kf0, s0);
      s1 = wmma_bf16(qf0, kf1, s1);
      kf0 = load_b_frag(Kp + (size_t)j0 * HDIM + 32, HDIM);
      kf1 = load_b_frag(Kp + (size_t)(j0 + 16) * HDIM + 32, HDIM);
      s0 = wmma_bf16(qf1, kf0, s0);
      s1 = wmma_bf16(qf1, kf1, s1);
    }
    // online softmax, per row (row = r + 8*hl within tile; 16 lanes share a row)
#pragma unroll
    for (int r = 0; r < 8; ++r) {
      const int row = q0 + r + 8 * hl;
      float a0 = s0[r] * 0.125f;                  // 1/sqrt(64)
      float a1 = s1[r] * 0.125f;
      if (j0 + ln > row)      a0 = -1e30f;
      if (j0 + 16 + ln > row) a1 = -1e30f;
      float mx = fmaxf(a0, a1);
#pragma unroll
      for (int msk = 8; msk >= 1; msk >>= 1) mx = fmaxf(mx, __shfl_xor(mx, msk, 32));
      const float mnew = fmaxf(mi[r], mx);
      const float p0 = __expf(a0 - mnew);
      const float p1 = __expf(a1 - mnew);
      const float sc = __expf(mi[r] - mnew);
      float ps = p0 + p1;
#pragma unroll
      for (int msk = 8; msk >= 1; msk >>= 1) ps += __shfl_xor(ps, msk, 32);
      li[r] = li[r] * sc + ps;
      mi[r] = mnew;
#pragma unroll
      for (int nt = 0; nt < 4; ++nt) oacc[nt][r] = oacc[nt][r] * sc;
      sP[wave][r + 8 * hl][ln]      = f2bf(p0);
      sP[wave][r + 8 * hl][16 + ln] = f2bf(p1);
    }
    // per-wave LDS RAW: waves diverge in trip count, so no barrier -- wait on DScnt instead
    asm volatile("s_wait_dscnt 0" ::: "memory");
    v16bf pf = load_a_frag(&sP[wave][0][0], 40);
#pragma unroll
    for (int nt = 0; nt < 4; ++nt) {
      v16bf vf = load_b_frag(Vt + (size_t)nt * 16 * CAPK + j0, CAPK);
      oacc[nt] = wmma_bf16(pf, vf, oacc[nt]);
    }
  }

#pragma unroll
  for (int r = 0; r < 8; ++r) {
    const float inv = 1.0f / li[r];
    const int row = q0 + r + 8 * hl;
    const size_t base = ((size_t)(b * CAPK + row)) * DMODEL + hh * HDIM;
#pragma unroll
    for (int nt = 0; nt < 4; ++nt)
      ob[base + nt * 16 + ln] = f2bf(oacc[nt][r] * inv);
  }
}

// ---------------- 9) SwiGLU gate: g = silu(u1) * u2 (bf16) ----------------
__global__ __launch_bounds__(256) void silu_mul_kernel(const float* __restrict__ u1,
                                                       const float* __restrict__ u2,
                                                       unsigned short* __restrict__ g,
                                                       int n) {
  const int i = blockIdx.x * 256 + threadIdx.x;
  if (i >= n) return;
  const float a = u1[i];
  const float s = a / (1.f + __expf(-a));
  g[i] = f2bf(s * u2[i]);
}

// ---------------- 10) out = x (vectorized copy) ----------------
__global__ __launch_bounds__(256) void copy4_kernel(const float4* __restrict__ in,
                                                    float4* __restrict__ out, int n) {
  const int i = blockIdx.x * 256 + threadIdx.x;
  if (i < n) out[i] = in[i];
}

// ---------------- 11) scatter processed tokens back ----------------
__global__ __launch_bounds__(256) void scatter_kernel(const float* __restrict__ xsel,
                                                      const float* __restrict__ xattn,
                                                      const float* __restrict__ xff,
                                                      const int* __restrict__ idx,
                                                      float* __restrict__ out) {
  const int i = blockIdx.x * 256 + threadIdx.x;   // < NTOK*DMODEL
  if (i >= NTOK * DMODEL) return;
  const int tok = i >> 10, c = i & 1023;
  const int b = tok >> 9, j = tok & 511;
  const int t = idx[b * CAPK + j];
  out[((size_t)b * TSEQ + t) * DMODEL + c] = xsel[i] + xattn[i] + xff[i];
}

// ================= host orchestration =================
extern "C" void kernel_launch(void* const* d_in, const int* in_sizes, int n_in,
                              void* d_out, int out_size, void* d_ws, size_t ws_size,
                              hipStream_t stream) {
  (void)in_sizes; (void)n_in; (void)out_size; (void)ws_size;
  const float* x        = (const float*)d_in[0];
  const float* w_router = (const float*)d_in[1];
  const float* W_qkv    = (const float*)d_in[2];
  const float* W_out    = (const float*)d_in[3];
  const float* g1       = (const float*)d_in[4];
  const float* g2       = (const float*)d_in[5];
  const float* W1       = (const float*)d_in[6];
  const float* W2       = (const float*)d_in[7];
  const float* W3       = (const float*)d_in[8];
  float* out = (float*)d_out;

  size_t off = 0;
  auto alloc = [&](size_t bytes) -> void* {
    void* p = (char*)d_ws + off;
    off = (off + bytes + 255) & ~(size_t)255;
    return p;
  };
  float*          scores  = (float*)alloc((size_t)NB * TSEQ * 4);
  int*            idx     = (int*)alloc((size_t)NB * CAPK * 4);
  float*          xsel    = (float*)alloc((size_t)NTOK * DMODEL * 4);
  unsigned short* hbuf    = (unsigned short*)alloc((size_t)NTOK * DMODEL * 2);
  unsigned short* wqkv_bf = (unsigned short*)alloc((size_t)D3 * DMODEL * 2);
  unsigned short* wout_bf = (unsigned short*)alloc((size_t)DMODEL * DMODEL * 2);
  unsigned short* w1p     = (unsigned short*)alloc((size_t)DFFP * DMODEL * 2);
  unsigned short* w2p     = (unsigned short*)alloc((size_t)DFFP * DMODEL * 2);
  unsigned short* w3p     = (unsigned short*)alloc((size_t)DMODEL * DFFP * 2);
  float*          qkv     = (float*)alloc((size_t)NTOK * D3 * 4);
  unsigned short* qbf     = (unsigned short*)alloc((size_t)NB * NHEAD * CAPK * HDIM * 2);
  unsigned short* kbf     = (unsigned short*)alloc((size_t)NB * NHEAD * CAPK * HDIM * 2);
  unsigned short* vtbf    = (unsigned short*)alloc((size_t)NB * NHEAD * HDIM * CAPK * 2);
  unsigned short* obf     = (unsigned short*)alloc((size_t)NTOK * DMODEL * 2);
  float*          xattn   = (float*)alloc((size_t)NTOK * DMODEL * 4);
  unsigned short* h2      = (unsigned short*)alloc((size_t)NTOK * DMODEL * 2);
  float*          u1      = (float*)alloc((size_t)NTOK * DFFP * 4);
  float*          u2      = (float*)alloc((size_t)NTOK * DFFP * 4);
  unsigned short* gbuf    = (unsigned short*)alloc((size_t)NTOK * DFFP * 2);
  float*          xff     = (float*)alloc((size_t)NTOK * DMODEL * 4);

  // out = x
  const int n4 = NB * TSEQ * DMODEL / 4;
  copy4_kernel<<<n4 / 256, 256, 0, stream>>>((const float4*)x, (float4*)out, n4);

  // routing
  router_kernel<<<NB * TSEQ / 8, 256, 0, stream>>>(x, w_router, scores);
  topk_kernel<<<NB, 256, 0, stream>>>(scores, idx);
  gather_norm_kernel<<<dim3(CAPK, NB), 256, 0, stream>>>(x, idx, g1, xsel, hbuf);

  // weight conversions (fp32 -> bf16, with DFF->DFFP zero padding)
  cvt_pad_kernel<<<(D3 * DMODEL + 255) / 256, 256, 0, stream>>>(W_qkv, wqkv_bf, D3, DMODEL, D3 * DMODEL, DMODEL);
  cvt_pad_kernel<<<(DMODEL * DMODEL + 255) / 256, 256, 0, stream>>>(W_out, wout_bf, DMODEL, DMODEL, DMODEL * DMODEL, DMODEL);
  cvt_pad_kernel<<<(DFFP * DMODEL + 255) / 256, 256, 0, stream>>>(W1, w1p, DFF, DMODEL, DFFP * DMODEL, DMODEL);
  cvt_pad_kernel<<<(DFFP * DMODEL + 255) / 256, 256, 0, stream>>>(W2, w2p, DFF, DMODEL, DFFP * DMODEL, DMODEL);
  cvt_pad_kernel<<<(DMODEL * DFFP + 255) / 256, 256, 0, stream>>>(W3, w3p, DMODEL, DFF, DMODEL * DFFP, DFFP);

  // QKV projection: [2048,1024] x [3072,1024]^T
  gemm_bf16_nt<<<dim3(D3 / 128, NTOK / 128), 256, 0, stream>>>(hbuf, wqkv_bf, qkv, NTOK, D3, DMODEL);
  qkv_split_kernel<<<(NTOK * D3 + 255) / 256, 256, 0, stream>>>(qkv, qbf, kbf, vtbf);

  // attention
  attn_kernel<<<dim3(CAPK / 64, NHEAD, NB), 128, 0, stream>>>(qbf, kbf, vtbf, obf);

  // output projection
  gemm_bf16_nt<<<dim3(DMODEL / 128, NTOK / 128), 256, 0, stream>>>(obf, wout_bf, xattn, NTOK, DMODEL, DMODEL);

  // second norm + SwiGLU FFN
  norm2_kernel<<<dim3(CAPK, NB), 256, 0, stream>>>(xsel, xattn, g2, h2);
  gemm_bf16_nt<<<dim3(DFFP / 128, NTOK / 128), 256, 0, stream>>>(h2, w1p, u1, NTOK, DFFP, DMODEL);
  gemm_bf16_nt<<<dim3(DFFP / 128, NTOK / 128), 256, 0, stream>>>(h2, w2p, u2, NTOK, DFFP, DMODEL);
  silu_mul_kernel<<<(NTOK * DFFP + 255) / 256, 256, 0, stream>>>(u1, u2, gbuf, NTOK * DFFP);
  gemm_bf16_nt<<<dim3(DMODEL / 128, NTOK / 128), 256, 0, stream>>>(gbuf, w3p, xff, NTOK, DMODEL, DFFP);

  // combine + scatter
  scatter_kernel<<<(NTOK * DMODEL + 255) / 256, 256, 0, stream>>>(xsel, xattn, xff, idx, out);
}
